// Allegro_Module_2886218023148
// MI455X (gfx1250) — matrix operand order repaired and checked by reading, not compiled, not run on hardware
//
#include <hip/hip_runtime.h>
#include <hip/hip_bf16.h>

// ---------------------------------------------------------------------------
// Allegro block for MI455X (gfx1250): fused per-16-edge-tile pipeline using
// v_wmma_f32_16x16x32_f16 for every GEMM (proj, TP channel mixing, both MLPs).
// f32 inputs/outputs, f16 WMMA operands, f32 accumulation.
// ---------------------------------------------------------------------------

typedef __attribute__((ext_vector_type(16))) _Float16 v16h;
typedef __attribute__((ext_vector_type(8)))  float    v8f;

#define NATOMS   10000
#define SF_W     0.17677669529663687f   // 1/sqrt(32)
#define INV_S3   0.57735026918962576f   // 1/sqrt(3)
#define INV_S6   0.40824829046386302f   // 1/sqrt(6)

static __device__ __forceinline__ v8f zero8() {
  v8f z = {0.f,0.f,0.f,0.f,0.f,0.f,0.f,0.f};
  return z;
}

static __device__ __forceinline__ v8f wmma16(v16h a, v16h b, v8f c) {
  // D = A(16x32 f16) * B(32x16 f16) + C(16x16 f32)
  return __builtin_amdgcn_wmma_f32_16x16x32_f16(false, a, false, b, (short)0, c, false, false);
}

// A fragment (16x32) from row-major f32 (global or LDS), converting to f16.
// ISA layout: lanes 0-15 row M=lane, K chunks [ko..ko+7] and [ko+16..ko+23],
// ko = (lane>>4)*8.
static __device__ __forceinline__ v16h fragA_f32(const float* base, int ld, int lane) {
  int row = lane & 15;
  int ko  = (lane >> 4) << 3;
  const float* p = base + (size_t)row * ld + ko;
  v16h a;
#pragma unroll
  for (int i = 0; i < 8; ++i) { a[i] = (_Float16)p[i]; a[i + 8] = (_Float16)p[i + 16]; }
  return a;
}

// B fragment (32x16) from weights stored transposed [N][K] in f16.
// Lane holds column N=lane&15, K chunks [ko..ko+7] and [ko+16..ko+23], ko=(lane>>4)*8.
static __device__ __forceinline__ v16h fragB_f16(const _Float16* wT, int ldk, int lane,
                                                 int ntile, int kbase) {
  int col = lane & 15;
  int ko  = (lane >> 4) << 3;
  const _Float16* p = wT + (size_t)(ntile * 16 + col) * ldk + kbase + ko;
  v16h b;
#pragma unroll
  for (int i = 0; i < 8; ++i) { b[i] = p[i]; b[i + 8] = p[i + 16]; }
  return b;
}

// D fragment store: col = lane&15, rows (lane>>4)*8 .. +7
static __device__ __forceinline__ void storeD(float* base, int ld, int lane, v8f d) {
  int col = lane & 15;
  int rb  = (lane >> 4) << 3;
#pragma unroll
  for (int r = 0; r < 8; ++r) base[(size_t)(rb + r) * ld + col] = d[r];
}

static __device__ __forceinline__ v8f silu8(v8f x) {
  v8f y;
#pragma unroll
  for (int i = 0; i < 8; ++i) y[i] = x[i] / (1.0f + __expf(-x[i]));
  return y;
}

// Scatter env_weight(basis, env_w)*SF into per-atom env via f32 atomics.
// dreg[t] = D-fragments of env_w columns t*16.. (64 cols = 32 muls x {w0,w1}).
static __device__ __forceinline__ void scatter_env(const v8f* dreg, const int* ctr,
                                                   const float* bas, float* env, int lane) {
  int col = lane & 15;
  int rb  = (lane >> 4) << 3;
#pragma unroll
  for (int t = 0; t < 4; ++t) {
    int c = t * 16 + col;   // 0..63 within env_w window
    int u = c >> 1;
    int odd = c & 1;
#pragma unroll
    for (int r = 0; r < 8; ++r) {
      int e = rb + r;
      float val = dreg[t][r] * SF_W;
      float* dst = env + (size_t)ctr[e] * 128 + u * 4;
      if (!odd) {
        atomicAdd(dst + 0, val * bas[e * 4 + 0]);
      } else {
        atomicAdd(dst + 1, val * bas[e * 4 + 1]);
        atomicAdd(dst + 2, val * bas[e * 4 + 2]);
        atomicAdd(dst + 3, val * bas[e * 4 + 3]);
      }
    }
  }
}

// ---------------------------------------------------------------------------
// K0 helpers: zero env buffers, transpose+convert weights [K][N]f32 -> [N][K]f16
// ---------------------------------------------------------------------------
__global__ void zero_f32(float* p, int n) {
  int i = blockIdx.x * blockDim.x + threadIdx.x;
  int stride = gridDim.x * blockDim.x;
  for (; i < n; i += stride) p[i] = 0.f;
}

__global__ void transpose_to_f16(const float* __restrict__ src, _Float16* __restrict__ dst,
                                 int K, int N) {
  int i = blockIdx.x * blockDim.x + threadIdx.x;
  if (i < K * N) {
    int n = i / K, k = i - n * K;
    dst[i] = (_Float16)src[(size_t)k * N + n];
  }
}

// ---------------------------------------------------------------------------
// K1: proj = edge_emb @ proj_W ; out[:,0:128] ; scatter env1
// 4 waves/block, each wave = one 16-edge tile.
// ---------------------------------------------------------------------------
__global__ __launch_bounds__(128) void k1_proj(
    const float* __restrict__ eemb, const float* __restrict__ basis,
    const int* __restrict__ center, const _Float16* __restrict__ projWt,
    float* __restrict__ out, float* __restrict__ env1, int ntiles) {
  __shared__ int   sctr[4][16];
  __shared__ float sbas[4][64];
  int wave = threadIdx.x >> 5, lane = threadIdx.x & 31;
  int tile = blockIdx.x * 4 + wave;
  if (tile >= ntiles) return;
  int e0 = tile * 16;
  int* ctr = sctr[wave];
  float* bas = sbas[wave];
  if (lane < 16) {
    ctr[lane] = center[e0 + lane];
#pragma unroll
    for (int j = 0; j < 4; ++j) bas[lane * 4 + j] = basis[(size_t)(e0 + lane) * 4 + j];
  }
  __builtin_amdgcn_wave_barrier();

  v16h a0 = fragA_f32(eemb + (size_t)e0 * 64 + 0, 64, lane);
  v16h a1 = fragA_f32(eemb + (size_t)e0 * 64 + 32, 64, lane);
  v8f dreg[4];
#pragma unroll
  for (int nt = 0; nt < 12; ++nt) {
    v8f d = zero8();
    d = wmma16(a0, fragB_f16(projWt, 64, lane, nt, 0), d);
    d = wmma16(a1, fragB_f16(projWt, 64, lane, nt, 32), d);
    if (nt < 8) storeD(out + (size_t)e0 * 384 + nt * 16, 384, lane, d);
    else        dreg[nt - 8] = d;
  }
  scatter_env(dreg, ctr, bas, env1, lane);
}

// ---------------------------------------------------------------------------
// K2: gather env1 -> t00/t110 -> scalars (WMMA mixing) -> MLP0 -> out[:,128:256]
//     + scatter env2. 4 waves/block.
// ---------------------------------------------------------------------------
__global__ __launch_bounds__(128) void k2_mid(
    const float* __restrict__ tfeat, const float* __restrict__ basis,
    const int* __restrict__ center,
    const _Float16* __restrict__ w00t, const _Float16* __restrict__ w110t,
    const _Float16* __restrict__ l0w1t, const _Float16* __restrict__ l0w2t,
    const float* __restrict__ env1, float* __restrict__ env2,
    float* __restrict__ out, int ntiles) {
  __shared__ float sP[4][512], sQ[4][512], sS[4][512], sH[4][2048];
  __shared__ float sbas[4][64];
  __shared__ int   sctr[4][16];
  int wave = threadIdx.x >> 5, lane = threadIdx.x & 31;
  int tile = blockIdx.x * 4 + wave;
  if (tile >= ntiles) return;
  int e0 = tile * 16;
  float *P = sP[wave], *Q = sQ[wave], *S = sS[wave], *H = sH[wave];
  float* bas = sbas[wave];
  int* ctr = sctr[wave];
  if (lane < 16) {
    ctr[lane] = center[e0 + lane];
#pragma unroll
    for (int j = 0; j < 4; ++j) bas[lane * 4 + j] = basis[(size_t)(e0 + lane) * 4 + j];
  }
  __builtin_amdgcn_wave_barrier();

  // t00[e,u] = f0*g0 ; t110[e,u] = (f.g)_vec / sqrt3
#pragma unroll 4
  for (int i = 0; i < 16; ++i) {
    int p = i * 32 + lane, e = p >> 5, u = p & 31;
    const float4 f = *(const float4*)(tfeat + (size_t)(e0 + e) * 128 + u * 4);
    const float4 g = *(const float4*)(env1 + (size_t)ctr[e] * 128 + u * 4);
    P[p] = f.x * g.x;
    Q[p] = (f.y * g.y + f.z * g.z + f.w * g.w) * INV_S3;
  }
  __builtin_amdgcn_wave_barrier();

  // scalars = t00 @ W00 + t110 @ W110   (16x32 @ 32x32)
  {
    v16h aP = fragA_f32(P, 32, lane);
    v16h aQ = fragA_f32(Q, 32, lane);
#pragma unroll
    for (int nt = 0; nt < 2; ++nt) {
      v8f d = zero8();
      d = wmma16(aP, fragB_f16(w00t, 32, lane, nt, 0), d);
      d = wmma16(aQ, fragB_f16(w110t, 32, lane, nt, 0), d);
      storeD(S + nt * 16, 32, lane, d);
    }
  }
  __builtin_amdgcn_wave_barrier();

  // hidden = silu([proj128 | scalars] @ W1)   K=160
  {
    v16h xf[5];
#pragma unroll
    for (int ks = 0; ks < 4; ++ks)
      xf[ks] = fragA_f32(out + (size_t)e0 * 384 + ks * 32, 384, lane);
    xf[4] = fragA_f32(S, 32, lane);
#pragma unroll
    for (int nt = 0; nt < 8; ++nt) {
      v8f d = zero8();
#pragma unroll
      for (int ks = 0; ks < 5; ++ks)
        d = wmma16(xf[ks], fragB_f16(l0w1t, 160, lane, nt, ks * 32), d);
      storeD(H + nt * 16, 128, lane, silu8(d));
    }
  }
  __builtin_amdgcn_wave_barrier();

  // lat0 = hidden @ W2   (128 -> 192); cols 0..127 -> out, 128..191 -> env_w2
  v8f dreg[4];
  {
    v16h hf[4];
#pragma unroll
    for (int ks = 0; ks < 4; ++ks) hf[ks] = fragA_f32(H + ks * 32, 128, lane);
#pragma unroll
    for (int nt = 0; nt < 12; ++nt) {
      v8f d = zero8();
#pragma unroll
      for (int ks = 0; ks < 4; ++ks)
        d = wmma16(hf[ks], fragB_f16(l0w2t, 128, lane, nt, ks * 32), d);
      if (nt < 8) storeD(out + (size_t)e0 * 384 + 128 + nt * 16, 384, lane, d);
      else        dreg[nt - 8] = d;
    }
  }
  scatter_env(dreg, ctr, bas, env2, lane);
}

// ---------------------------------------------------------------------------
// K3: recompute full tf (tp0), tp1 contraction with env2, MLP1 -> out[:,256:384]
// 2 waves/block (LDS budget), each wave one 16-edge tile.
// ---------------------------------------------------------------------------
__global__ __launch_bounds__(64) void k3_final(
    const float* __restrict__ tfeat, const int* __restrict__ center,
    const _Float16* __restrict__ w00t,  const _Float16* __restrict__ w110t,
    const _Float16* __restrict__ w011t, const _Float16* __restrict__ w101t,
    const _Float16* __restrict__ w111t,
    const _Float16* __restrict__ t1w00t, const _Float16* __restrict__ t1w110t,
    const _Float16* __restrict__ l1w1t,  const _Float16* __restrict__ l1w2t,
    const float* __restrict__ env1, const float* __restrict__ env2,
    float* __restrict__ out, int ntiles) {
  __shared__ float sF[2][2048], sG[2][2048];             // feat tile (reused as hidden), gathered env1
  __shared__ float sP[2][512], sQ[2][512], sR[2][512];   // t-work tiles
  __shared__ float sA110[2][512], sT200[2][512], sS2[2][512];
  __shared__ int   sctr[2][16];
  int wave = threadIdx.x >> 5, lane = threadIdx.x & 31;
  int tile = blockIdx.x * 2 + wave;
  if (tile >= ntiles) return;
  int e0 = tile * 16;
  float *F = sF[wave], *G = sG[wave];
  float *P = sP[wave], *Q = sQ[wave], *R = sR[wave];
  float *A110 = sA110[wave], *T200 = sT200[wave], *S2 = sS2[wave];
  int* ctr = sctr[wave];
  if (lane < 16) ctr[lane] = center[e0 + lane];
  __builtin_amdgcn_wave_barrier();

  // Stage tensor_features tile and gathered env1 tile (each 16x32x4 f32)
#pragma unroll 4
  for (int i = 0; i < 16; ++i) {
    int p = i * 32 + lane;
    *(float4*)(F + p * 4) = *(const float4*)(tfeat + (size_t)e0 * 128 + p * 4);
  }
#pragma unroll 4
  for (int i = 0; i < 16; ++i) {
    int p = i * 32 + lane, e = p >> 5, u = p & 31;
    *(float4*)(G + p * 4) = *(const float4*)(env1 + (size_t)ctr[e] * 128 + u * 4);
  }
  __builtin_amdgcn_wave_barrier();

  // --- tf0 = t00 @ W00 + t110 @ W110, then t2_00 = tf0 * g2_0 ---
#pragma unroll 4
  for (int i = 0; i < 16; ++i) {
    int p = i * 32 + lane;
    const float* f = F + p * 4;
    const float* g = G + p * 4;
    P[p] = f[0] * g[0];
    Q[p] = (f[1] * g[1] + f[2] * g[2] + f[3] * g[3]) * INV_S3;
    A110[p] = 0.f;
  }
  __builtin_amdgcn_wave_barrier();
  {
    v16h aP = fragA_f32(P, 32, lane);
    v16h aQ = fragA_f32(Q, 32, lane);
#pragma unroll
    for (int nt = 0; nt < 2; ++nt) {
      v8f d = zero8();
      d = wmma16(aP, fragB_f16(w00t, 32, lane, nt, 0), d);
      d = wmma16(aQ, fragB_f16(w110t, 32, lane, nt, 0), d);
      int v = nt * 16 + (lane & 15);
      int rb = (lane >> 4) << 3;
#pragma unroll
      for (int r = 0; r < 8; ++r) {
        int e = rb + r;
        float g20 = env2[(size_t)ctr[e] * 128 + v * 4 + 0];
        T200[e * 32 + v] = d[r] * g20;
      }
    }
  }
  __builtin_amdgcn_wave_barrier();

  // --- tf_k for k=0..2 (t011/t101/t111 paths), accumulate t2_110 ---
#pragma unroll
  for (int k = 0; k < 3; ++k) {
    int ia = (k + 1) % 3, ib = (k + 2) % 3;
#pragma unroll 4
    for (int i = 0; i < 16; ++i) {
      int p = i * 32 + lane;
      const float* f = F + p * 4;
      const float* g = G + p * 4;
      P[p] = f[0] * g[1 + k] * INV_S3;                                 // t011_k
      Q[p] = f[1 + k] * g[0] * INV_S3;                                 // t101_k
      R[p] = (f[1 + ia] * g[1 + ib] - f[1 + ib] * g[1 + ia]) * INV_S6; // t111_k
    }
    __builtin_amdgcn_wave_barrier();
    v16h aP = fragA_f32(P, 32, lane);
    v16h aQ = fragA_f32(Q, 32, lane);
    v16h aR = fragA_f32(R, 32, lane);
#pragma unroll
    for (int nt = 0; nt < 2; ++nt) {
      v8f d = zero8();
      d = wmma16(aP, fragB_f16(w011t, 32, lane, nt, 0), d);
      d = wmma16(aQ, fragB_f16(w101t, 32, lane, nt, 0), d);
      d = wmma16(aR, fragB_f16(w111t, 32, lane, nt, 0), d);
      int v = nt * 16 + (lane & 15);
      int rb = (lane >> 4) << 3;
#pragma unroll
      for (int r = 0; r < 8; ++r) {
        int e = rb + r;
        float g2k = env2[(size_t)ctr[e] * 128 + v * 4 + 1 + k];
        A110[e * 32 + v] += d[r] * g2k * INV_S3;
      }
    }
    __builtin_amdgcn_wave_barrier();
  }

  // --- scalars2 = t2_00 @ tp1_W00 + t2_110 @ tp1_W110 ---
  {
    v16h aT = fragA_f32(T200, 32, lane);
    v16h aA = fragA_f32(A110, 32, lane);
#pragma unroll
    for (int nt = 0; nt < 2; ++nt) {
      v8f d = zero8();
      d = wmma16(aT, fragB_f16(t1w00t, 32, lane, nt, 0), d);
      d = wmma16(aA, fragB_f16(t1w110t, 32, lane, nt, 0), d);
      storeD(S2 + nt * 16, 32, lane, d);
    }
  }
  __builtin_amdgcn_wave_barrier();

  // --- hidden = silu([proj128 | lat0_128 | scalars2] @ W1), K=288 ---
  {
    v16h xf[9];
#pragma unroll
    for (int ks = 0; ks < 8; ++ks)
      xf[ks] = fragA_f32(out + (size_t)e0 * 384 + ks * 32, 384, lane);
    xf[8] = fragA_f32(S2, 32, lane);
#pragma unroll
    for (int nt = 0; nt < 8; ++nt) {
      v8f d = zero8();
#pragma unroll
      for (int ks = 0; ks < 9; ++ks)
        d = wmma16(xf[ks], fragB_f16(l1w1t, 288, lane, nt, ks * 32), d);
      storeD(F + nt * 16, 128, lane, silu8(d));   // F reused as hidden tile
    }
  }
  __builtin_amdgcn_wave_barrier();

  // --- lat1 = hidden @ W2 (128 -> 128) -> out cols 256..383 ---
  {
    v16h hf[4];
#pragma unroll
    for (int ks = 0; ks < 4; ++ks) hf[ks] = fragA_f32(F + ks * 32, 128, lane);
#pragma unroll
    for (int nt = 0; nt < 8; ++nt) {
      v8f d = zero8();
#pragma unroll
      for (int ks = 0; ks < 4; ++ks)
        d = wmma16(hf[ks], fragB_f16(l1w2t, 128, lane, nt, ks * 32), d);
      storeD(out + (size_t)e0 * 384 + 256 + nt * 16, 384, lane, d);
    }
  }
}

// ---------------------------------------------------------------------------
extern "C" void kernel_launch(void* const* d_in, const int* in_sizes, int n_in,
                              void* d_out, int out_size, void* d_ws, size_t ws_size,
                              hipStream_t stream) {
  const int*   edge_index = (const int*)  d_in[0];   // (2,E): row 0 = edge_center
  const float* basis      = (const float*)d_in[1];   // (E,4)
  const float* tfeat      = (const float*)d_in[2];   // (E,32,4)
  const float* eemb       = (const float*)d_in[3];   // (E,64)
  const float* proj_W     = (const float*)d_in[5];   // (64,192)
  const float* lat0_W1    = (const float*)d_in[6];   // (160,128)
  const float* lat0_W2    = (const float*)d_in[7];   // (128,192)
  const float* lat1_W1    = (const float*)d_in[8];   // (288,128)
  const float* lat1_W2    = (const float*)d_in[9];   // (128,128)
  const float* tp0_w00    = (const float*)d_in[10];
  const float* tp0_w110   = (const float*)d_in[11];
  const float* tp0_w011   = (const float*)d_in[12];
  const float* tp0_w101   = (const float*)d_in[13];
  const float* tp0_w111   = (const float*)d_in[14];
  const float* tp1_w00    = (const float*)d_in[15];
  const float* tp1_w110   = (const float*)d_in[16];
  float* out = (float*)d_out;

  const int E = in_sizes[3] / 64;
  const int ntiles = E / 16;

  // Workspace layout: env1, env2 (f32, L2-resident), then f16 transposed weights
  float* env1 = (float*)d_ws;
  float* env2 = env1 + (size_t)NATOMS * 128;
  _Float16* wp = (_Float16*)(env2 + (size_t)NATOMS * 128);
  _Float16* projWt  = wp; wp += 192 * 64;
  _Float16* l0w1t   = wp; wp += 128 * 160;
  _Float16* l0w2t   = wp; wp += 192 * 128;
  _Float16* l1w1t   = wp; wp += 128 * 288;
  _Float16* l1w2t   = wp; wp += 128 * 128;
  _Float16* w00t    = wp; wp += 1024;
  _Float16* w110t   = wp; wp += 1024;
  _Float16* w011t   = wp; wp += 1024;
  _Float16* w101t   = wp; wp += 1024;
  _Float16* w111t   = wp; wp += 1024;
  _Float16* t1w00t  = wp; wp += 1024;
  _Float16* t1w110t = wp; wp += 1024;

  // K0: zero env accumulators (required every call), prep f16 weights
  zero_f32<<<2048, 256, 0, stream>>>(env1, 2 * NATOMS * 128);
  auto tr = [&](const float* s, _Float16* d, int K, int N) {
    int n = K * N;
    transpose_to_f16<<<(n + 255) / 256, 256, 0, stream>>>(s, d, K, N);
  };
  tr(proj_W,   projWt,  64, 192);
  tr(lat0_W1,  l0w1t,  160, 128);
  tr(lat0_W2,  l0w2t,  128, 192);
  tr(lat1_W1,  l1w1t,  288, 128);
  tr(lat1_W2,  l1w2t,  128, 128);
  tr(tp0_w00,  w00t,    32, 32);
  tr(tp0_w110, w110t,   32, 32);
  tr(tp0_w011, w011t,   32, 32);
  tr(tp0_w101, w101t,   32, 32);
  tr(tp0_w111, w111t,   32, 32);
  tr(tp1_w00,  t1w00t,  32, 32);
  tr(tp1_w110, t1w110t, 32, 32);

  k1_proj<<<(ntiles + 3) / 4, 128, 0, stream>>>(
      eemb, basis, edge_index, projWt, out, env1, ntiles);
  k2_mid<<<(ntiles + 3) / 4, 128, 0, stream>>>(
      tfeat, basis, edge_index, w00t, w110t, l0w1t, l0w2t, env1, env2, out, ntiles);
  k3_final<<<(ntiles + 1) / 2, 64, 0, stream>>>(
      tfeat, edge_index, w00t, w110t, w011t, w101t, w111t,
      t1w00t, t1w110t, l1w1t, l1w2t, env1, env2, out, ntiles);
}
